// ProposedModel3_90159953478283
// MI455X (gfx1250) — compile-verified, matrix operand order checked by reading, not run on hardware
//
#include <hip/hip_runtime.h>
#include <hip/hip_bf16.h>

typedef unsigned int uint;
typedef __attribute__((ext_vector_type(16))) __bf16 v16bf;
typedef __attribute__((ext_vector_type(8)))  float  v8f;

// ---------- helpers ----------
__device__ __forceinline__ unsigned short f2bf(float f) {
    uint x; __builtin_memcpy(&x, &f, 4);
    uint r = x + 0x7FFFu + ((x >> 16) & 1u);      // round-to-nearest-even
    return (unsigned short)(r >> 16);
}
__device__ __forceinline__ float bf2f(unsigned short u) {
    uint x = ((uint)u) << 16; float f; __builtin_memcpy(&f, &x, 4); return f;
}
__device__ __forceinline__ v16bf load16(const uint* p) {
    union { uint4 q[2]; v16bf v; } u;
    u.q[0] = *(const uint4*)(p);
    u.q[1] = *(const uint4*)(p + 8);
    return u.v;
}
__device__ __forceinline__ float sigm(float x) { return 1.f / (1.f + __expf(-x)); }
__device__ __forceinline__ float lse2(float a, float b) {
    float mx = fmaxf(a, b), mn = fminf(a, b);
    return mx + log1pf(__expf(mn - mx));
}

// ---------- tiny utility kernels ----------
__global__ void k_zero(float* p, int n) {
    int i = blockIdx.x * 256 + threadIdx.x;
    if (i < n) p[i] = 0.f;
}
__global__ void k_init_e(float* e, const float* fc_b) {
    int i = blockIdx.x * 256 + threadIdx.x;   // 131072 elements
    e[i] = fc_b[i & 1];
}
__global__ void k_cvt_bf16(const float* src, unsigned short* dst, int n) {
    int i = blockIdx.x * 256 + threadIdx.x;
    if (i < n) dst[i] = f2bf(src[i]);
}
__global__ void k_bias_sum(const float* a, const float* b, float* dst, int n) {
    int i = blockIdx.x * 256 + threadIdx.x;
    if (i < n) dst[i] = a[i] + b[i];
}

// ---------- BatchNorm statistics: column sums of 65536 x 768 ----------
__global__ __launch_bounds__(256) void k_bn_stats(const float* __restrict__ x,
                                                  float* __restrict__ sum,
                                                  float* __restrict__ sq) {
    int tid = threadIdx.x;
    size_t row0 = (size_t)blockIdx.x * 256;
    float s[3] = {0.f, 0.f, 0.f}, q[3] = {0.f, 0.f, 0.f};
    for (int r = 0; r < 256; ++r) {
        const float* p = x + (row0 + r) * 768;
#pragma unroll
        for (int ff = 0; ff < 3; ++ff) {
            float v = p[tid + ff * 256];
            s[ff] += v; q[ff] += v * v;
        }
    }
#pragma unroll
    for (int ff = 0; ff < 3; ++ff) {
        atomicAdd(&sum[tid + ff * 256], s[ff]);
        atomicAdd(&sq [tid + ff * 256], q[ff]);
    }
}

__global__ void k_bn_finalize(const float* sum, const float* sq,
                              const float* gamma, const float* beta,
                              float* scale, float* shift) {
    int i = blockIdx.x * 256 + threadIdx.x;
    if (i >= 768) return;
    const float invN = 1.f / 65536.f;
    float mu = sum[i] * invN;
    float var = sq[i] * invN - mu * mu;
    float sc = gamma[i] * rsqrtf(var + 1e-5f);
    scale[i] = sc;
    shift[i] = beta[i] - mu * sc;
}

// ---------- fused WMMA GEMM + LSTM-cell-at-t0 (h0=c0=0 -> f gate dead) ----------
// Block: 64 rows x 128 h-units x 1 direction; 8 waves. Wave w owns j-subtile
// w*16..w*16+15 and accumulates 4 M-subtiles x {i,g,o} = 12 f32 16x16 tiles,
// so each B fragment (L2) feeds 4 WMMAs (B traffic /4 vs the M=16 version).
// fc_mode: fuse the 512->2 FC through an LDS h-tile + f32 atomics into e.
__global__ __launch_bounds__(256) void k_gemm_lstm(
    const void* __restrict__ Aptr, int a_is_fp32,
    const float* __restrict__ scale, const float* __restrict__ shift,
    int K,
    const unsigned short* __restrict__ W,   // [2048][K] bf16 row-major
    const float* __restrict__ bias,         // [2048] = b_ih + b_hh
    unsigned short* __restrict__ Hout,      // [65536][512] bf16 (or null)
    int fc_mode, const float* __restrict__ fcw, float* __restrict__ e)
{
    __shared__ unsigned short sA[64 * 776];   // padded A tile (bank-conflict free)
    __shared__ unsigned short sH[64 * 128];   // h tile for fused FC

    const int tid  = threadIdx.x;
    const int wave = tid >> 5, lane = tid & 31;
    const int n0 = blockIdx.x * 64;
    const int jc = blockIdx.y;
    const int d  = blockIdx.z;
    const int Kp = K + 8, Kp2 = Kp >> 1, K2 = K >> 1;

    // ---- stage A tile (with fused BN apply for layer 0), vectorized ----
    if (a_is_fp32) {
        const float4* A4 = (const float4*)((const float*)Aptr + (size_t)n0 * K);
        const int K4 = K >> 2;                        // float4s per row
        for (int i = tid; i < 64 * K4; i += 256) {
            int r = i / K4, c4 = (i - r * K4) << 2;
            float4 v = A4[i];
            float f0 = v.x * scale[c4 + 0] + shift[c4 + 0];
            float f1 = v.y * scale[c4 + 1] + shift[c4 + 1];
            float f2 = v.z * scale[c4 + 2] + shift[c4 + 2];
            float f3 = v.w * scale[c4 + 3] + shift[c4 + 3];
            uint2 pk;
            pk.x = (uint)f2bf(f0) | ((uint)f2bf(f1) << 16);
            pk.y = (uint)f2bf(f2) | ((uint)f2bf(f3) << 16);
            *(uint2*)&sA[r * Kp + c4] = pk;
        }
    } else {
        const uint4* A4 = (const uint4*)((const unsigned short*)Aptr + (size_t)n0 * K);
        const int K8 = K >> 3;                        // uint4s per row (8 bf16)
        for (int i = tid; i < 64 * K8; i += 256) {
            int r = i / K8, c = (i - r * K8) << 2;    // uint column
            *(uint4*)((uint*)sA + r * Kp2 + c) = A4[i];
        }
    }
    __syncthreads();

    const int jbase = jc * 128 + wave * 16;
    const int nlane = lane & 15, kh = lane >> 4;
    const int rowI = d * 1024 + 0   + jbase + nlane;   // i gate
    const int rowG = d * 1024 + 512 + jbase + nlane;   // g gate
    const int rowO = d * 1024 + 768 + jbase + nlane;   // o gate

    v8f acc[4][3];
#pragma unroll
    for (int mt = 0; mt < 4; ++mt)
#pragma unroll
        for (int g = 0; g < 3; ++g) acc[mt][g] = (v8f){};

    const uint* s32 = (const uint*)sA;
    const uint* w32 = (const uint*)W;

    for (int k0 = 0; k0 < K; k0 += 32) {
        int koff = (k0 + kh * 8) >> 1;                 // u32 offset within a row
        v16bf bI = load16(w32 + (size_t)rowI * K2 + koff);
        v16bf bG = load16(w32 + (size_t)rowG * K2 + koff);
        v16bf bO = load16(w32 + (size_t)rowO * K2 + koff);
#pragma unroll
        for (int mt = 0; mt < 4; ++mt) {
            v16bf a = load16(s32 + (mt * 16 + nlane) * Kp2 + koff);
            acc[mt][0] = __builtin_amdgcn_wmma_f32_16x16x32_bf16(false, a, false, bI, (short)0, acc[mt][0], false, false);
            acc[mt][1] = __builtin_amdgcn_wmma_f32_16x16x32_bf16(false, a, false, bG, (short)0, acc[mt][1], false, false);
            acc[mt][2] = __builtin_amdgcn_wmma_f32_16x16x32_bf16(false, a, false, bO, (short)0, acc[mt][2], false, false);
        }
    }

    // ---- in-register LSTM epilogue: h = sig(o)*tanh(sig(i)*tanh(g)) ----
    const float bI_ = bias[d * 1024 + 0   + jbase + nlane];
    const float bG_ = bias[d * 1024 + 512 + jbase + nlane];
    const float bO_ = bias[d * 1024 + 768 + jbase + nlane];
#pragma unroll
    for (int mt = 0; mt < 4; ++mt) {
#pragma unroll
        for (int r = 0; r < 8; ++r) {
            int mm = mt * 16 + r + kh * 8;             // C layout: lanes>=16 hold M=r+8
            float iv = acc[mt][0][r] + bI_;
            float gv = acc[mt][1][r] + bG_;
            float ov = acc[mt][2][r] + bO_;
            float cc = sigm(iv) * tanhf(gv);
            float hv = sigm(ov) * tanhf(cc);
            unsigned short hb = f2bf(hv);
            if (!fc_mode)
                Hout[(size_t)(n0 + mm) * 512 + d * 256 + jbase + nlane] = hb;
            else
                sH[mm * 128 + wave * 16 + nlane] = hb;
        }
    }

    if (fc_mode) {
        __syncthreads();
        if (tid < 128) {                               // 64 rows x 2 classes
            int row = tid >> 1, c = tid & 1;
            const float* fw = fcw + c * 512 + d * 256 + jc * 128;
            float s = 0.f;
            for (int j = 0; j < 128; ++j) s += bf2f(sH[row * 128 + j]) * fw[j];
            atomicAdd(&e[(size_t)(n0 + row) * 2 + c], s);
        }
    }
}

// ---------- CRF NLL: parallel log-semiring reduction of 2x2 matrices ----------
__global__ __launch_bounds__(1024) void k_crf(const float* __restrict__ e,
                                              const int* __restrict__ tags,
                                              const float* __restrict__ start,
                                              const float* __restrict__ endv,
                                              const float* __restrict__ trans,
                                              float* __restrict__ loss)
{
    __shared__ float sM[1024][4];
    __shared__ float sN[1024];
    const int tid = threadIdx.x;
    const float T00 = trans[0], T01 = trans[1], T10 = trans[2], T11 = trans[3];

    // segment of step-matrices M_t[a][b] = trans[a][b] + e[t][b], t = 1..65535
    int t0 = 1 + tid * 64;
    int t1 = t0 + 64; if (t1 > 65536) t1 = 65536;
    float e0 = e[t0 * 2], e1 = e[t0 * 2 + 1];
    float m00 = T00 + e0, m01 = T01 + e1, m10 = T10 + e0, m11 = T11 + e1;
    for (int t = t0 + 1; t < t1; ++t) {
        e0 = e[t * 2]; e1 = e[t * 2 + 1];
        float n00 = T00 + e0, n01 = T01 + e1, n10 = T10 + e0, n11 = T11 + e1;
        float r00 = lse2(m00 + n00, m01 + n10);
        float r01 = lse2(m00 + n01, m01 + n11);
        float r10 = lse2(m10 + n00, m11 + n10);
        float r11 = lse2(m10 + n01, m11 + n11);
        m00 = r00; m01 = r01; m10 = r10; m11 = r11;
    }

    // numerator partial over n in [tid*64, tid*64+64)
    float num = 0.f;
    int nc = tid * 64;
    for (int n = nc; n < nc + 64; ++n) {
        int tg = tags[n];
        num += e[n * 2 + tg];
        if (n > 0) num += trans[tags[n - 1] * 2 + tg];
    }

    sM[tid][0] = m00; sM[tid][1] = m01; sM[tid][2] = m10; sM[tid][3] = m11;
    sN[tid] = num;
    __syncthreads();

    for (int s = 512; s >= 1; s >>= 1) {
        if (tid < s) {
            float L00 = sM[tid][0], L01 = sM[tid][1], L10 = sM[tid][2], L11 = sM[tid][3];
            float R00 = sM[tid+s][0], R01 = sM[tid+s][1], R10 = sM[tid+s][2], R11 = sM[tid+s][3];
            sM[tid][0] = lse2(L00 + R00, L01 + R10);
            sM[tid][1] = lse2(L00 + R01, L01 + R11);
            sM[tid][2] = lse2(L10 + R00, L11 + R10);
            sM[tid][3] = lse2(L10 + R01, L11 + R11);
            sN[tid] += sN[tid + s];
        }
        __syncthreads();
    }

    if (tid == 0) {
        float a0 = start[0] + e[0], a1 = start[1] + e[1];
        float f0 = lse2(a0 + sM[0][0], a1 + sM[0][2]);
        float f1 = lse2(a0 + sM[0][1], a1 + sM[0][3]);
        float den = lse2(f0 + endv[0], f1 + endv[1]);
        float numt = sN[0] + start[tags[0]] + endv[tags[65535]];
        loss[0] = -(numt - den);
    }
}

// ---------- host-side orchestration ----------
extern "C" void kernel_launch(void* const* d_in, const int* in_sizes, int n_in,
                              void* d_out, int out_size, void* d_ws, size_t ws_size,
                              hipStream_t stream) {
    const float* x        = (const float*)d_in[0];
    const float* bn_gamma = (const float*)d_in[1];
    const float* bn_beta  = (const float*)d_in[2];
    const float* w_ih_l0  = (const float*)d_in[3];
    const float* b_ih_l0  = (const float*)d_in[5];
    const float* b_hh_l0  = (const float*)d_in[6];
    const float* w_ih_l1  = (const float*)d_in[7];
    const float* b_ih_l1  = (const float*)d_in[9];
    const float* b_hh_l1  = (const float*)d_in[10];
    const float* fc_w     = (const float*)d_in[11];
    const float* fc_b     = (const float*)d_in[12];
    const float* crf_start= (const float*)d_in[13];
    const float* crf_end  = (const float*)d_in[14];
    const float* crf_trans= (const float*)d_in[15];
    const int*   labels   = (const int*)d_in[16];

    char* ws = (char*)d_ws;
    float* w_sum   = (float*)(ws + 0);         // 768
    float* w_sq    = (float*)(ws + 3072);      // 768
    float* w_scale = (float*)(ws + 6144);      // 768
    float* w_shift = (float*)(ws + 9216);      // 768
    float* w_bias0 = (float*)(ws + 12288);     // 2048
    float* w_bias1 = (float*)(ws + 20480);     // 2048
    unsigned short* w_W0 = (unsigned short*)(ws + 32768);                       // 2x1024x768 bf16
    unsigned short* w_W1 = (unsigned short*)(ws + 32768 + 3145728);             // 2x1024x512 bf16
    unsigned short* w_H1 = (unsigned short*)(ws + 32768 + 3145728 + 2097152);   // 65536x512 bf16

    float* out = (float*)d_out;   // [0]=loss, [1..131072]=logits (e)
    float* e   = out + 1;

    k_zero    <<<6,   256, 0, stream>>>(w_sum, 1536);             // sum+sq contiguous
    k_init_e  <<<512, 256, 0, stream>>>(e, fc_b);
    k_bn_stats<<<256, 256, 0, stream>>>(x, w_sum, w_sq);
    k_bn_finalize<<<3, 256, 0, stream>>>(w_sum, w_sq, bn_gamma, bn_beta, w_scale, w_shift);
    k_cvt_bf16<<<6144, 256, 0, stream>>>(w_ih_l0, w_W0, 1572864);
    k_cvt_bf16<<<4096, 256, 0, stream>>>(w_ih_l1, w_W1, 1048576);
    k_bias_sum<<<8,   256, 0, stream>>>(b_ih_l0, b_hh_l0, w_bias0, 2048);
    k_bias_sum<<<8,   256, 0, stream>>>(b_ih_l1, b_hh_l1, w_bias1, 2048);

    // Layer 0: A = BN(x) fused in-loader (fp32 -> bf16), K=768, write h1_in
    k_gemm_lstm<<<dim3(1024, 2, 2), 256, 0, stream>>>(
        x, 1, w_scale, w_shift, 768, w_W0, w_bias0, w_H1, 0, nullptr, nullptr);
    // Layer 1: A = h1_in bf16, K=512, fused FC epilogue into e
    k_gemm_lstm<<<dim3(1024, 2, 2), 256, 0, stream>>>(
        w_H1, 0, nullptr, nullptr, 512, w_W1, w_bias1, nullptr, 1, fc_w, e);

    k_crf<<<1, 1024, 0, stream>>>(e, labels, crf_start, crf_end, crf_trans, out);
}
